// Model_49933289783861
// MI455X (gfx1250) — matrix operand order repaired and checked by reading, not compiled, not run on hardware
//
#include <hip/hip_runtime.h>
#include <hip/hip_fp16.h>

// MI455X / gfx1250. Memory-bound workload (~0.8GB HBM traffic @23.3TB/s ~ 35us);
// all GEMMs via v_wmma_f32_16x16x32_f16 with f16 hi/lo split for ~fp32 accuracy.

typedef __attribute__((ext_vector_type(16))) _Float16 v16h;
typedef __attribute__((ext_vector_type(8)))  _Float16 v8h;
typedef __attribute__((ext_vector_type(8)))  float    v8f;

#define NROWS 262144
#define PDIM  256
#define HDIM  256
#define NCLS  10

// ---- workspace byte offsets ----
#define GRAM_OFF   0u          // 65536 f32
#define COL_OFF    262144u     // 256 f32
#define SCAL_OFF   263168u     // 16 f32: 0 ce, 1 wsum, 2 w2sum, 3 b1abs, 4 b2abs, 5 correct
#define B1FH_OFF   264192u     // 65536 halfs (b1 hi fragments)
#define B1FL_OFF   395264u     // 65536 halfs (b1 lo fragments)
#define B2FH_OFF   526336u     // 4096 halfs  (b2^T hi fragments, classes padded to 16)
#define B2FL_OFF   534528u     // 4096 halfs
// total ws used: 542720 bytes

static __device__ __forceinline__ v8f wmma16(v16h a, v16h b, v8f c) {
  // D = A(16x32 f16) * B(32x16 f16) + C(16x16 f32)
  return __builtin_amdgcn_wmma_f32_16x16x32_f16(false, a, false, b, (short)0, c,
                                                false, false);
}
static __device__ __forceinline__ void fsplit(float x, _Float16& hi, _Float16& lo) {
  hi = (_Float16)x;
  lo = (_Float16)(x - (float)hi);
}
// A-fragment K index for element e (0..15), lane group g (0..1)  [ISA 7.12.2]
static __device__ __forceinline__ int kmapA(int e, int g) {
  return (e & 7) + 8 * g + 16 * (e >> 3);
}
static __device__ __forceinline__ v8f zerov8() {
  v8f z = {0.f, 0.f, 0.f, 0.f, 0.f, 0.f, 0.f, 0.f};
  return z;
}

// ---------------------------------------------------------------------------
// Kernel 0: zero accumulator region of workspace
// ---------------------------------------------------------------------------
__global__ __launch_bounds__(256) void zero_ws(float* wsf) {
  unsigned i = blockIdx.x * 256u + threadIdx.x;
  if (i < 65808u) wsf[i] = 0.0f;  // gram(65536) + col(256) + scalars(16)
}

// ---------------------------------------------------------------------------
// Kernel 1: pre-swizzle b1 into WMMA B-fragment layout (hi/lo f16),
//           b2^T into WMMA A-fragment layout (classes padded to 16),
//           and accumulate |b1|, |b2| sums.
// grid = 17 blocks x 256.
// ---------------------------------------------------------------------------
__global__ __launch_bounds__(256) void prep_b(const float* __restrict__ b1,
                                              const float* __restrict__ b2,
                                              char* __restrict__ ws) {
  __shared__ float s_abs;
  if (threadIdx.x == 0) s_abs = 0.f;
  __syncthreads();

  const int wave = threadIdx.x >> 5, lane = threadIdx.x & 31;
  const int g = lane >> 4, l = lane & 15;
  _Float16* b1h = (_Float16*)(ws + B1FH_OFF);
  _Float16* b1l = (_Float16*)(ws + B1FL_OFF);
  _Float16* b2h = (_Float16*)(ws + B2FH_OFF);
  _Float16* b2l = (_Float16*)(ws + B2FL_OFF);
  float* scal = (float*)(ws + SCAL_OFF);

  float aacc = 0.f;
  if (blockIdx.x < 16) {
    // b1 B-fragments: frag f = kt*16+nt; lane group g covers K = kt*32+kmapA(e,g)
    const int f = blockIdx.x * 8 + wave;
    const int kt = f >> 4, nt = f & 15;
#pragma unroll
    for (int e = 0; e < 16; ++e) {
      const int k = kt * 32 + kmapA(e, g);
      const int n = nt * 16 + l;
      const float v = b1[k * HDIM + n];
      aacc += fabsf(v);
      _Float16 hi, lo;
      fsplit(v, hi, lo);
      const unsigned idx = ((unsigned)(f * 32 + lane) << 4) + e;
      b1h[idx] = hi;
      b1l[idx] = lo;
    }
  } else {
    // b2^T A-fragments: 8 kt frags; A[m=class][k] = b2[k][m], pad classes 10..15
    const int kt = wave;
#pragma unroll
    for (int e = 0; e < 16; ++e) {
      const int kk = kt * 32 + 16 * g + e;
      const float v = (l < NCLS) ? b2[kk * NCLS + l] : 0.0f;
      aacc += fabsf(v);
      _Float16 hi, lo;
      fsplit(v, hi, lo);
      const unsigned idx = ((unsigned)(kt * 32 + lane) << 4) + e;
      b2h[idx] = hi;
      b2l[idx] = lo;
    }
  }
  atomicAdd(&s_abs, aacc);
  __syncthreads();
  if (threadIdx.x == 0) atomicAdd(&scal[(blockIdx.x < 16) ? 3 : 4], s_abs);
}

// ---------------------------------------------------------------------------
// Kernel 2: fused MLP forward: hdn = relu(x@b1), dropout, logits = hm@b2,
//           per-sample CE + argmax accuracy, accumulated atomically.
// grid = NROWS/64 blocks x 128 threads (4 waves, 16 rows each).
// ---------------------------------------------------------------------------
__global__ __launch_bounds__(128) void mlp_main(const float* __restrict__ x,
                                                const int* __restrict__ y,
                                                const float* __restrict__ du,
                                                char* __restrict__ ws) {
  __shared__ _Float16 s_xh[64][40];       // x tile hi (row stride 40 -> 16B aligned runs)
  __shared__ _Float16 s_xl[64][40];       // x tile lo
  __shared__ _Float16 s_hm[4][16][264];   // masked hidden per wave (f16)
  __shared__ float s_ce, s_corr;

  const int tid = threadIdx.x;
  const int wave = tid >> 5, lane = tid & 31, g = lane >> 4, l = lane & 15;
  if (tid == 0) { s_ce = 0.f; s_corr = 0.f; }
  const long wgRow0 = (long)blockIdx.x * 64;

  const _Float16* b1h = (const _Float16*)(ws + B1FH_OFF);
  const _Float16* b1l = (const _Float16*)(ws + B1FL_OFF);
  const _Float16* b2h = (const _Float16*)(ws + B2FH_OFF);
  const _Float16* b2l = (const _Float16*)(ws + B2FL_OFF);
  float* scal = (float*)(ws + SCAL_OFF);

  v8f acc[16];
#pragma unroll
  for (int i = 0; i < 16; ++i) acc[i] = zerov8();

  for (int kt = 0; kt < 8; ++kt) {
    const int k0 = kt * 32;
    __syncthreads();
    // stage 64x32 slab of x into LDS (coalesced), split to f16 hi/lo
#pragma unroll
    for (int p = 0; p < 16; ++p) {
      const int id = p * 128 + tid;
      const int row = id >> 5, col = id & 31;
      const float xv = x[(wgRow0 + row) * PDIM + k0 + col];
      if (kt < 7) __builtin_prefetch(&x[(wgRow0 + row) * PDIM + k0 + 32 + col], 0, 1);
      _Float16 hi, lo;
      fsplit(xv, hi, lo);
      s_xh[row][col] = hi;
      s_xl[row][col] = lo;
    }
    __syncthreads();
    // A fragment for this wave's 16 rows (runs of 8 contiguous halfs)
    const int arow = wave * 16 + l;
    v8h a0h = *(const v8h*)&s_xh[arow][8 * g];
    v8h a1h = *(const v8h*)&s_xh[arow][16 + 8 * g];
    v8h a0l = *(const v8h*)&s_xl[arow][8 * g];
    v8h a1l = *(const v8h*)&s_xl[arow][16 + 8 * g];
    v16h ah = __builtin_shufflevector(a0h, a1h, 0, 1, 2, 3, 4, 5, 6, 7, 8, 9, 10, 11, 12, 13, 14, 15);
    v16h al = __builtin_shufflevector(a0l, a1l, 0, 1, 2, 3, 4, 5, 6, 7, 8, 9, 10, 11, 12, 13, 14, 15);
#pragma unroll
    for (int nt = 0; nt < 16; ++nt) {
      const unsigned fo = ((unsigned)((kt * 16 + nt) * 32 + lane)) << 4;
      const v16h bh = *(const v16h*)(b1h + fo);
      const v16h bl = *(const v16h*)(b1l + fo);
      acc[nt] = wmma16(ah, bh, acc[nt]);   // hi*hi
      acc[nt] = wmma16(ah, bl, acc[nt]);   // hi*lo
      acc[nt] = wmma16(al, bh, acc[nt]);   // lo*hi
    }
  }

  // relu + inverted dropout; stash masked hidden (f16) for the logits pass
#pragma unroll
  for (int nt = 0; nt < 16; ++nt) {
#pragma unroll
    for (int v = 0; v < 8; ++v) {
      const int rl = v + 8 * g;                  // C-layout: M = v + 8g
      const long row = wgRow0 + wave * 16 + rl;
      const int col = nt * 16 + l;               // N = lane
      float hv = acc[nt][v];
      hv = hv > 0.f ? hv : 0.f;
      const float u = du[row * HDIM + col];
      const float hm = (u > 0.3f) ? hv * (1.0f / 0.7f) : 0.0f;
      s_hm[wave][rl][col] = (_Float16)hm;
    }
  }

  // logits^T = b2^T (A) x hm^T (B): lane l ends up owning sample l's logits
  v8f accL = zerov8();
  for (int kt = 0; kt < 8; ++kt) {
    const unsigned fo = ((unsigned)(kt * 32 + lane)) << 4;
    const v16h a2h = *(const v16h*)(b2h + fo);
    const v16h a2l = *(const v16h*)(b2l + fo);
    const v8h* bp = (const v8h*)&s_hm[wave][l][kt * 32 + 16 * g];
    v16h bb = __builtin_shufflevector(bp[0], bp[1], 0, 1, 2, 3, 4, 5, 6, 7, 8, 9, 10, 11, 12, 13, 14, 15);
    accL = wmma16(a2h, bb, accL);
    accL = wmma16(a2l, bb, accL);
  }
  // classes 8,9 live in lane l+16 (M = v + 8g); pull them over
  const int src = (lane & 15) + 16;
  const float c8 = __shfl(accL[0], src, 32);
  const float c9 = __shfl(accL[1], src, 32);
  if (lane < 16) {
    const long row = wgRow0 + wave * 16 + lane;
    float lg[10];
#pragma unroll
    for (int c = 0; c < 8; ++c) lg[c] = accL[c];
    lg[8] = c8;
    lg[9] = c9;
    float mx = lg[0];
    int am = 0;
#pragma unroll
    for (int c = 1; c < 10; ++c)
      if (lg[c] > mx) { mx = lg[c]; am = c; }
    float se = 0.f;
#pragma unroll
    for (int c = 0; c < 10; ++c) se += __expf(lg[c] - mx);
    const float lse = __logf(se) + mx;
    const int yi = y[row];
    float ly = lg[0];
#pragma unroll
    for (int c = 1; c < 10; ++c)
      if (yi == c) ly = lg[c];
    atomicAdd(&s_ce, lse - ly);
    if (am == yi) atomicAdd(&s_corr, 1.0f);
  }
  __syncthreads();
  if (tid == 0) {
    atomicAdd(&scal[0], s_ce);
    atomicAdd(&scal[5], s_corr);
  }
}

// ---------------------------------------------------------------------------
// Kernel 3: gram = x^T diag(w^2) x = (Swx)^T(Swx)/S^2, col = x^T w^2,
//           plus sum(w^2), sum(w^4).
// grid = 64 blocks x 256 threads; each block owns 4096 rows of x.
// ---------------------------------------------------------------------------
__global__ __launch_bounds__(256) void gram_kernel(const float* __restrict__ x,
                                                   const float* __restrict__ w,
                                                   char* __restrict__ ws) {
  __shared__ _Float16 t_h[256][40];  // (S*w*x)^T tile hi: [col][k], stride 40 halfs
  __shared__ _Float16 t_l[256][40];
  __shared__ float s_w[32];
  __shared__ float s_ws, s_w2s;

  const float S = 512.0f, invS2 = 1.0f / (512.0f * 512.0f);
  const int tid = threadIdx.x;
  const int wave = tid >> 5, lane = tid & 31, g = lane >> 4, l = lane & 15;
  if (tid == 0) { s_ws = 0.f; s_w2s = 0.f; }
  const long chunk0 = (long)blockIdx.x * 4096;

  float* gram = (float*)(ws + GRAM_OFF);
  float* col = (float*)(ws + COL_OFF);
  float* scal = (float*)(ws + SCAL_OFF);

  v8f acc[32];  // wave owns m-tiles {wave, wave+8} x all 16 n-tiles
#pragma unroll
  for (int i = 0; i < 32; ++i) acc[i] = zerov8();

  float colacc = 0.f, wsp = 0.f, w2p = 0.f;

  for (int kt = 0; kt < 128; ++kt) {
    const long base = chunk0 + kt * 32;
    __syncthreads();
    if (tid < 32) {
      const float wv = w[base + tid];
      s_w[tid] = wv;
      const float w2 = wv * wv;
      wsp += w2;
      w2p += w2 * w2;
    }
    __syncthreads();
    // stage transposed tile: t[col][k] = S * w_k * x[base+k][col]; col accumulates x^T w^2
#pragma unroll 4
    for (int k = 0; k < 32; ++k) {
      const float xv = x[(base + k) * PDIM + tid];
      if (kt < 127) __builtin_prefetch(&x[(base + 32 + k) * PDIM + tid], 0, 1);
      const float wk = s_w[k];
      colacc += wk * wk * xv;
      _Float16 hi, lo;
      fsplit(S * wk * xv, hi, lo);
      t_h[tid][k] = hi;
      t_l[tid][k] = lo;
    }
    __syncthreads();
    // A fragments for the 2 owned m-tiles (A[m][k] = t[m][k], K contiguous)
    v16h ah[2], al[2];
#pragma unroll
    for (int m = 0; m < 2; ++m) {
      const int ac = (wave + 8 * m) * 16 + l;
      v8h r0h = *(const v8h*)&t_h[ac][16 * g];
      v8h r1h = *(const v8h*)&t_h[ac][16 * g + 8];
      v8h r0l = *(const v8h*)&t_l[ac][16 * g];
      v8h r1l = *(const v8h*)&t_l[ac][16 * g + 8];
      ah[m] = __builtin_shufflevector(r0h, r1h, 0, 1, 2, 3, 4, 5, 6, 7, 8, 9, 10, 11, 12, 13, 14, 15);
      al[m] = __builtin_shufflevector(r0l, r1l, 0, 1, 2, 3, 4, 5, 6, 7, 8, 9, 10, 11, 12, 13, 14, 15);
    }
#pragma unroll
    for (int nt = 0; nt < 16; ++nt) {
      const int jc = nt * 16 + l;  // B[k][j]: lane j holds column j, K contiguous
      const v8h* bph = (const v8h*)&t_h[jc][16 * g];
      const v8h* bpl = (const v8h*)&t_l[jc][16 * g];
      v16h bh = __builtin_shufflevector(bph[0], bph[1], 0, 1, 2, 3, 4, 5, 6, 7, 8, 9, 10, 11, 12, 13, 14, 15);
      v16h bl = __builtin_shufflevector(bpl[0], bpl[1], 0, 1, 2, 3, 4, 5, 6, 7, 8, 9, 10, 11, 12, 13, 14, 15);
#pragma unroll
      for (int m = 0; m < 2; ++m) {
        acc[m * 16 + nt] = wmma16(ah[m], bh, acc[m * 16 + nt]);
        acc[m * 16 + nt] = wmma16(ah[m], bl, acc[m * 16 + nt]);
        acc[m * 16 + nt] = wmma16(al[m], bh, acc[m * 16 + nt]);
      }
    }
  }

  // write out partial gram (undo S^2), col, and w moments
#pragma unroll
  for (int m = 0; m < 2; ++m) {
#pragma unroll
    for (int nt = 0; nt < 16; ++nt) {
#pragma unroll
      for (int v = 0; v < 8; ++v) {
        const int i = (wave + 8 * m) * 16 + v + 8 * g;
        const int j = nt * 16 + l;
        atomicAdd(&gram[i * PDIM + j], acc[m * 16 + nt][v] * invS2);
      }
    }
  }
  atomicAdd(&col[tid], colacc);
  if (tid < 32) {
    atomicAdd(&s_ws, wsp);
    atomicAdd(&s_w2s, w2p);
  }
  __syncthreads();
  if (tid == 0) {
    atomicAdd(&scal[1], s_ws);
    atomicAdd(&scal[2], s_w2s);
  }
}

// ---------------------------------------------------------------------------
// Kernel 4: finalize: bm regularizer + loss assembly + accuracy.
// ---------------------------------------------------------------------------
__global__ __launch_bounds__(256) void finalize(char* __restrict__ ws,
                                                float* __restrict__ out) {
  __shared__ float s_col[256];
  __shared__ float s_bm;
  const int t = threadIdx.x;
  float* gram = (float*)(ws + GRAM_OFF);
  float* col = (float*)(ws + COL_OFF);
  float* scal = (float*)(ws + SCAL_OFF);
  if (t == 0) s_bm = 0.f;
  s_col[t] = col[t];
  __syncthreads();
  const float ci = s_col[t];
  float bm = 0.f;
  for (int j = 0; j < PDIM; ++j) {
    if (j == t) continue;
    const float d = gram[t * PDIM + j] - ci * s_col[j];
    bm += d * d;
  }
  atomicAdd(&s_bm, bm);
  __syncthreads();
  if (t == 0) {
    const float ce = scal[0], wsum = scal[1], w2sum = scal[2];
    const float b1a = scal[3], b2a = scal[4], corr = scal[5];
    float loss = ce * wsum;
    loss += 0.01f * (b1a + b2a);     // L1
    loss += 0.001f * s_bm;           // L2
    loss += 0.1f * w2sum;            // L3
    loss += 100.0f * (wsum - 1.0f) * (wsum - 1.0f);  // L4
    out[0] = loss;
    out[1] = corr / (float)NROWS;
  }
}

// ---------------------------------------------------------------------------
extern "C" void kernel_launch(void* const* d_in, const int* in_sizes, int n_in,
                              void* d_out, int out_size, void* d_ws, size_t ws_size,
                              hipStream_t stream) {
  const float* x = (const float*)d_in[0];
  const int* y = (const int*)d_in[1];     // labels (int32 on device)
  const float* du = (const float*)d_in[2];
  const float* b1 = (const float*)d_in[3];
  const float* b2 = (const float*)d_in[4];
  const float* w = (const float*)d_in[5];
  float* out = (float*)d_out;
  char* ws = (char*)d_ws;

  zero_ws<<<258, 256, 0, stream>>>((float*)ws);
  prep_b<<<17, 256, 0, stream>>>(b1, b2, ws);
  mlp_main<<<NROWS / 64, 128, 0, stream>>>(x, y, du, ws);
  gram_kernel<<<64, 256, 0, stream>>>(x, w, ws);
  finalize<<<1, 256, 0, stream>>>(ws, out);
}
